// egret_ppi_84542136254806
// MI455X (gfx1250) — compile-verified, matrix operand order checked by reading, not run on hardware
//
#include <hip/hip_runtime.h>
#include <hip/hip_bf16.h>
#include <stdint.h>

// ---------------------------------------------------------------------------
// EGRET-PPI pipeline for MI455X (gfx1250, wave32, WMMA bf16)
//   conv(T5) GEMM + biLSTM + dense GEMM + 2x GAT + head
// Dominant work (conv im2col GEMM, 67.6 GFLOP) runs on v_wmma_f32_16x16x32_bf16.
// ---------------------------------------------------------------------------

typedef __attribute__((ext_vector_type(16))) __bf16 v16bf;
typedef __attribute__((ext_vector_type(8)))  float  v8f;
union ABu { v16bf v; uint4 u[2]; };

#define B_    32
#define L_    768
#define T_    11
#define FD_   33
#define N_    (B_*L_)        // 24576 nodes
#define E_    (N_*20)        // 491520 edges
#define DT5_  1024
#define KCONV_ (21*1024)     // 21504
#define DCONV_ 64
#define HL_   16
#define DZ_   96
#define LPAD_ (L_+20)

__device__ __forceinline__ unsigned short f2bf(float f) {
  unsigned u = __float_as_uint(f);
  u += 0x7FFFu + ((u >> 16) & 1u);      // round-to-nearest-even
  return (unsigned short)(u >> 16);
}
__device__ __forceinline__ float sigm(float x) { return 1.0f / (1.0f + __expf(-x)); }
// order-preserving uint encoding of float for atomicMax (memset-0 == -inf sentinel)
__device__ __forceinline__ unsigned fkey(float f) {
  unsigned u = __float_as_uint(f);
  return (u & 0x80000000u) ? ~u : (u | 0x80000000u);
}
__device__ __forceinline__ float unfkey(unsigned k) {
  unsigned u = (k & 0x80000000u) ? (k & 0x7FFFFFFFu) : ~k;
  return __uint_as_float(u);
}

// ---------------------------------------------------------------------- zero
__global__ void k_zero(unsigned int* p, long n) {
  long i = (long)blockIdx.x * 256 + threadIdx.x;
  if (i < n) p[i] = 0u;
}

// --------------------------------------------------- bf16 padded T5 buffer
__global__ void k_prep_pad(const float* __restrict__ t5, unsigned short* __restrict__ pad) {
  long i = (long)blockIdx.x * 256 + threadIdx.x;
  if (i >= (long)B_ * L_ * DT5_) return;
  int  c  = (int)(i % DT5_);
  long bl = i / DT5_;
  int  l  = (int)(bl % L_);
  int  b  = (int)(bl / L_);
  pad[((long)b * LPAD_ + l + 10) * DT5_ + c] = f2bf(t5[i]);
}

// ------------------------------------- conv weight -> fragment-ready B (bf16)
// B frag layout: [ntile][kb][lane][16], lane n=lane&15, khalf=lane>>4,
// elems 0..7 -> K = kb*32 + khalf*8 + j ; elems 8..15 -> K = kb*32+16+khalf*8+(j-8)
__global__ void k_prep_convfrag(const float* __restrict__ w, unsigned short* __restrict__ frag) {
  long idx = (long)blockIdx.x * 256 + threadIdx.x;
  const int nkb = KCONV_ / 32;
  if (idx >= (long)(DCONV_ / 16) * nkb * 512) return;
  int  j    = (int)(idx & 15);
  int  lane = (int)((idx >> 4) & 31);
  long rest = idx >> 9;
  int  kb   = (int)(rest % nkb);
  int  nt   = (int)(rest / nkb);
  int  n    = nt * 16 + (lane & 15);
  int  kh   = lane >> 4;
  int  k    = kb * 32 + (j < 8 ? kh * 8 + j : 16 + kh * 8 + (j - 8));
  int  c    = k & 1023, tap = k >> 10;       // K = tap*1024 + c
  frag[idx] = f2bf(w[((long)n * DT5_ + c) * 21 + tap]);   // conv_w[n][c][tap]
}

// generic weight -> fragment-ready B. nMajor: 1 => W[n*K+k], 0 => W[k*Nn+n]
__global__ void k_prep_wfrag(const float* __restrict__ w, unsigned short* __restrict__ frag,
                             int K, int Nn, int nMajor) {
  int nkb = K >> 5;
  long total = (long)(Nn / 16) * nkb * 512;
  long idx = (long)blockIdx.x * 256 + threadIdx.x;
  if (idx >= total) return;
  int  j    = (int)(idx & 15);
  int  lane = (int)((idx >> 4) & 31);
  long rest = idx >> 9;
  int  kb   = (int)(rest % nkb);
  int  nt   = (int)(rest / nkb);
  int  n    = nt * 16 + (lane & 15);
  int  kh   = lane >> 4;
  int  k    = kb * 32 + (j < 8 ? kh * 8 + j : 16 + kh * 8 + (j - 8));
  frag[idx] = f2bf(nMajor ? w[(long)n * K + k] : w[(long)k * Nn + n]);
}

// ------------------------------------------------------------ WMMA GEMM
// One 16x16 C tile per wave; Nn/16 waves per block; grid.x = M/16.
// mode 0: conv epilogue (bias + leaky + BN) -> z[:,32:96] f32 + bf16
// mode 1: dense epilogue (bias + relu)      -> z[:,0:32]  f32 + bf16
// mode 2: plain f32 -> outF (ldOut)
__global__ __launch_bounds__(128)
void k_gemm_bf16(const unsigned short* __restrict__ A, int lda, int batchL, int extraPerBatch,
                 const unsigned short* __restrict__ Bfrag, int K,
                 float* __restrict__ outF, int ldOut,
                 float* __restrict__ zOut, unsigned short* __restrict__ zbfOut, int zcol,
                 const float* __restrict__ bias,
                 const float* __restrict__ bn_g, const float* __restrict__ bn_b,
                 const float* __restrict__ bn_m, const float* __restrict__ bn_v,
                 int mode) {
  const int lane = threadIdx.x & 31;
  const int wave = threadIdx.x >> 5;
  const int mloc = lane & 15;
  const int kh   = lane >> 4;
  const int nkb  = K >> 5;

  const long r = (long)blockIdx.x * 16 + mloc;                 // A row this lane feeds
  const unsigned short* Arow = A + r * lda + (r / batchL) * (long)extraPerBatch;
  const unsigned short* bp   = Bfrag + ((long)wave * nkb * 32 + lane) * 16;

  v8f acc = {0.f, 0.f, 0.f, 0.f, 0.f, 0.f, 0.f, 0.f};
  for (int kb = 0; kb < nkb; ++kb) {
    const unsigned short* ap = Arow + kb * 32 + kh * 8;
    __builtin_prefetch((const void*)(ap + 256), 0, 1);         // global_prefetch
    ABu a, b;
    a.u[0] = *(const uint4*)(ap);
    a.u[1] = *(const uint4*)(ap + 16);
    b.u[0] = *(const uint4*)(bp);
    b.u[1] = *(const uint4*)(bp + 8);
    bp += 32 * 16;
    acc = __builtin_amdgcn_wmma_f32_16x16x32_bf16(false, a.v, false, b.v,
                                                  (short)0, acc, false, false);
  }

  const int n     = wave * 16 + (lane & 15);
  const long rb   = (long)blockIdx.x * 16 + (lane >> 4) * 8;   // C layout: lanes>=16 -> M+8
  for (int q = 0; q < 8; ++q) {
    float v = acc[q];
    long row = rb + q;
    if (mode == 0) {
      v += bias[n];
      v = v > 0.f ? v : 0.01f * v;
      v = (v - bn_m[n]) * rsqrtf(bn_v[n] + 1e-5f) * bn_g[n] + bn_b[n];
      zOut[row * DZ_ + zcol + n]   = v;
      zbfOut[row * DZ_ + zcol + n] = f2bf(v);
    } else if (mode == 1) {
      v += bias[n];
      v = fmaxf(v, 0.f);
      zOut[row * DZ_ + zcol + n]   = v;
      zbfOut[row * DZ_ + zcol + n] = f2bf(v);
    } else {
      outF[row * ldOut + n] = v;
    }
  }
}

// ----------------------------------------------------------------- biLSTM
__global__ __launch_bounds__(256)
void k_lstm(const float* __restrict__ feature, const float* __restrict__ ln_g,
            const float* __restrict__ ln_b,
            const float* __restrict__ wih_f, const float* __restrict__ whh_f,
            const float* __restrict__ bih_f, const float* __restrict__ bhh_f,
            const float* __restrict__ wih_b, const float* __restrict__ whh_b,
            const float* __restrict__ bih_b, const float* __restrict__ bhh_b,
            unsigned short* __restrict__ hcat) {
  __shared__ float s_wih[2][64][FD_];
  __shared__ float s_whh[2][64][HL_];
  __shared__ float s_b[2][64];
  const int tid = threadIdx.x;
  for (int i = tid; i < 64 * FD_; i += 256) {
    s_wih[0][i / FD_][i % FD_] = wih_f[i];
    s_wih[1][i / FD_][i % FD_] = wih_b[i];
  }
  for (int i = tid; i < 64 * HL_; i += 256) {
    s_whh[0][i / HL_][i % HL_] = whh_f[i];
    s_whh[1][i / HL_][i % HL_] = whh_b[i];
  }
  for (int i = tid; i < 64; i += 256) {
    s_b[0][i] = bih_f[i] + bhh_f[i];
    s_b[1][i] = bih_b[i] + bhh_b[i];
  }
  __syncthreads();

  long g = (long)blockIdx.x * 256 + tid;
  if (g >= 2L * N_) return;
  const int dir  = (g >= N_) ? 1 : 0;      // whole blocks share dir (N_%256==0)
  const long node = g - (long)dir * N_;
  const float* fb = feature + node * T_ * FD_;

  float h[HL_], c[HL_];
  for (int j = 0; j < HL_; ++j) { h[j] = 0.f; c[j] = 0.f; }

  for (int s = 0; s < T_; ++s) {
    const int t = dir ? (T_ - 1 - s) : s;
    float y[FD_];
    const float* fr = fb + t * FD_;
    float mu = 0.f;
    for (int i = 0; i < FD_; ++i) { y[i] = fr[i]; mu += y[i]; }
    mu *= (1.0f / FD_);
    float var = 0.f;
    for (int i = 0; i < FD_; ++i) { float d = y[i] - mu; var += d * d; }
    var *= (1.0f / FD_);
    const float inv = rsqrtf(var + 1e-6f);
    for (int i = 0; i < FD_; ++i) y[i] = (y[i] - mu) * inv * ln_g[i] + ln_b[i];

    float nh[HL_], nc[HL_];
    for (int j = 0; j < HL_; ++j) {
      float gi = s_b[dir][j], gf = s_b[dir][16 + j], gg = s_b[dir][32 + j], go = s_b[dir][48 + j];
      for (int i = 0; i < FD_; ++i) {
        const float yi = y[i];
        gi += s_wih[dir][j][i] * yi;
        gf += s_wih[dir][16 + j][i] * yi;
        gg += s_wih[dir][32 + j][i] * yi;
        go += s_wih[dir][48 + j][i] * yi;
      }
      for (int k = 0; k < HL_; ++k) {
        const float hk = h[k];
        gi += s_whh[dir][j][k] * hk;
        gf += s_whh[dir][16 + j][k] * hk;
        gg += s_whh[dir][32 + j][k] * hk;
        go += s_whh[dir][48 + j][k] * hk;
      }
      const float ci = sigm(gf) * c[j] + sigm(gi) * tanhf(gg);
      nc[j] = ci;
      nh[j] = sigm(go) * tanhf(ci);
    }
    for (int j = 0; j < HL_; ++j) { h[j] = nh[j]; c[j] = nc[j]; }
    unsigned short* o = hcat + node * (T_ * 32) + t * 32 + dir * HL_;
    for (int j = 0; j < HL_; ++j) o[j] = f2bf(h[j]);
  }
}

// ------------------------------------------------------------- GAT edges
__global__ void k_edge_score(const float* __restrict__ z, int D,
                             const int* __restrict__ src, const int* __restrict__ dst,
                             const float* __restrict__ ef, const float* __restrict__ we,
                             const float* __restrict__ a,
                             float* __restrict__ score, unsigned int* __restrict__ segmax) {
  int e = blockIdx.x * 256 + threadIdx.x;
  if (e >= E_) return;
  const int s = src[e], d = dst[e];
  const float* zs = z + (long)s * D;
  const float* zd = z + (long)d * D;
  const float* ep = ef + (long)e * 6;
  const float e0 = ep[0], e1 = ep[1], e2 = ep[2], e3 = ep[3], e4 = ep[4], e5 = ep[5];
  float sc = 0.f;
  for (int j = 0; j < D; ++j) {
    const float ez = e0 * we[j] + e1 * we[D + j] + e2 * we[2 * D + j] +
                     e3 * we[3 * D + j] + e4 * we[4 * D + j] + e5 * we[5 * D + j];
    sc += a[j] * zs[j] + a[D + j] * zd[j] + a[2 * D + j] * ez;
  }
  sc = sc > 0.f ? sc : 0.01f * sc;
  score[e] = sc;
  atomicMax(&segmax[d], fkey(sc));
}

__global__ void k_edge_norm(const float* __restrict__ score, const int* __restrict__ dst,
                            const unsigned int* __restrict__ segmax,
                            float* __restrict__ ex, float* __restrict__ segsum) {
  int e = blockIdx.x * 256 + threadIdx.x;
  if (e >= E_) return;
  const int d = dst[e];
  const float v = __expf(score[e] - unfkey(segmax[d]));
  ex[e] = v;
  atomicAdd(&segsum[d], v);
}

__global__ void k_edge_aggr(const float* __restrict__ z, int D,
                            const int* __restrict__ src, const int* __restrict__ dst,
                            const float* __restrict__ ef, const float* __restrict__ we,
                            const float* __restrict__ ex, const float* __restrict__ segsum,
                            float* __restrict__ h, float* __restrict__ alphaOut) {
  int e = blockIdx.x * 256 + threadIdx.x;
  if (e >= E_) return;
  const int s = src[e], d = dst[e];
  const float alpha = ex[e] / (segsum[d] + 1e-9f);
  if (alphaOut) alphaOut[e] = alpha;
  const float* zs = z + (long)s * D;
  const float* ep = ef + (long)e * 6;
  const float e0 = ep[0], e1 = ep[1], e2 = ep[2], e3 = ep[3], e4 = ep[4], e5 = ep[5];
  for (int j = 0; j < D; ++j) {
    const float ez = e0 * we[j] + e1 * we[D + j] + e2 * we[2 * D + j] +
                     e3 * we[3 * D + j] + e4 * we[4 * D + j] + e5 * we[5 * D + j];
    atomicAdd(&h[(long)d * D + j], alpha * (zs[j] + ez));
  }
}

__global__ void k_elu_bf(const float* __restrict__ h, unsigned short* __restrict__ hb, long n) {
  long i = (long)blockIdx.x * 256 + threadIdx.x;
  if (i >= n) return;
  float v = h[i];
  v = v > 0.f ? v : expm1f(v);
  hb[i] = f2bf(v);
}

__global__ void k_out_final(const float* __restrict__ h2, const float* __restrict__ z,
                            const float* __restrict__ w, const float* __restrict__ b,
                            float* __restrict__ out) {
  int i = blockIdx.x * 256 + threadIdx.x;
  if (i >= N_) return;
  float s = b[0];
  const float* hr = h2 + (long)i * 32;
  const float* zr = z + (long)i * DZ_;
  for (int j = 0; j < 32; ++j)  s += w[j] * hr[j];
  for (int j = 0; j < DZ_; ++j) s += w[32 + j] * zr[j];
  out[i] = sigm(s);
}

// ---------------------------------------------------------------------------
extern "C" void kernel_launch(void* const* d_in, const int* in_sizes, int n_in,
                              void* d_out, int out_size, void* d_ws, size_t ws_size,
                              hipStream_t stream) {
  const float* feature = (const float*)d_in[0];
  const float* t5      = (const float*)d_in[1];
  const float* edgef   = (const float*)d_in[2];
  const int*   src     = (const int*)d_in[3];
  const int*   dst     = (const int*)d_in[4];
  const float* conv_w  = (const float*)d_in[5];
  const float* conv_b  = (const float*)d_in[6];
  const float* bn_g    = (const float*)d_in[7];
  const float* bn_b    = (const float*)d_in[8];
  const float* bn_m    = (const float*)d_in[9];
  const float* bn_v    = (const float*)d_in[10];
  const float* ln_g    = (const float*)d_in[11];
  const float* ln_b    = (const float*)d_in[12];
  const float* wih_f   = (const float*)d_in[13];
  const float* whh_f   = (const float*)d_in[14];
  const float* bih_f   = (const float*)d_in[15];
  const float* bhh_f   = (const float*)d_in[16];
  const float* wih_b   = (const float*)d_in[17];
  const float* whh_b   = (const float*)d_in[18];
  const float* bih_b   = (const float*)d_in[19];
  const float* bhh_b   = (const float*)d_in[20];
  const float* dense_w = (const float*)d_in[21];
  const float* dense_b = (const float*)d_in[22];
  const float* gat_w1  = (const float*)d_in[23];
  const float* gat_we1 = (const float*)d_in[24];
  const float* gat_a1  = (const float*)d_in[25];
  const float* gat_w2  = (const float*)d_in[26];
  const float* gat_we2 = (const float*)d_in[27];
  const float* gat_a2  = (const float*)d_in[28];
  const float* out_w   = (const float*)d_in[29];
  const float* out_b   = (const float*)d_in[30];

  float* out    = (float*)d_out;   // N_ sigmoid outputs
  float* alpha2 = out + N_;        // E_ attention weights (2nd tuple element)

  char* ws = (char*)d_ws;
  size_t off = 0;
  auto alloc = [&](size_t bytes) -> char* {
    char* p = ws + off;
    off = (off + bytes + 255) & ~(size_t)255;
    return p;
  };
  unsigned short* t5pad    = (unsigned short*)alloc((size_t)B_ * LPAD_ * DT5_ * 2);
  unsigned short* convfrag = (unsigned short*)alloc((size_t)KCONV_ * DCONV_ * 2);
  unsigned short* densefrag= (unsigned short*)alloc((size_t)352 * 32 * 2);
  unsigned short* w1frag   = (unsigned short*)alloc((size_t)96 * 64 * 2);
  unsigned short* w2frag   = (unsigned short*)alloc((size_t)64 * 32 * 2);
  unsigned short* hcat     = (unsigned short*)alloc((size_t)N_ * 352 * 2);
  float*          z        = (float*)alloc((size_t)N_ * DZ_ * 4);
  unsigned short* zbf      = (unsigned short*)alloc((size_t)N_ * DZ_ * 2);
  float*          z1       = (float*)alloc((size_t)N_ * 64 * 4);
  float*          h1       = (float*)alloc((size_t)N_ * 64 * 4);
  unsigned short* h1bf     = (unsigned short*)alloc((size_t)N_ * 64 * 2);
  float*          z2       = (float*)alloc((size_t)N_ * 32 * 4);
  float*          h2       = (float*)alloc((size_t)N_ * 32 * 4);
  float*          score1   = (float*)alloc((size_t)E_ * 4);
  float*          ex1      = (float*)alloc((size_t)E_ * 4);
  unsigned int*   segmax1  = (unsigned int*)alloc((size_t)N_ * 4);
  float*          segsum1  = (float*)alloc((size_t)N_ * 4);
  float*          score2   = (float*)alloc((size_t)E_ * 4);
  float*          ex2      = (float*)alloc((size_t)E_ * 4);
  unsigned int*   segmax2  = (unsigned int*)alloc((size_t)N_ * 4);
  float*          segsum2  = (float*)alloc((size_t)N_ * 4);

  auto zero = [&](void* p, long words) {
    k_zero<<<(unsigned)((words + 255) / 256), 256, 0, stream>>>((unsigned int*)p, words);
  };
  zero(t5pad, (long)B_ * LPAD_ * DT5_ / 2);
  zero(h1, (long)N_ * 64);
  zero(h2, (long)N_ * 32);
  zero(segmax1, N_); zero(segsum1, N_);
  zero(segmax2, N_); zero(segsum2, N_);

  // ---- prep (bf16 conversion + fragment-ready weight swizzles) ----
  {
    long n = (long)B_ * L_ * DT5_;
    k_prep_pad<<<(unsigned)((n + 255) / 256), 256, 0, stream>>>(t5, t5pad);
  }
  {
    long n = (long)(DCONV_ / 16) * (KCONV_ / 32) * 512;
    k_prep_convfrag<<<(unsigned)((n + 255) / 256), 256, 0, stream>>>(conv_w, convfrag);
  }
  k_prep_wfrag<<<(2 * 11 * 512 + 255) / 256, 256, 0, stream>>>(dense_w, densefrag, 352, 32, 1);
  k_prep_wfrag<<<(4 * 3 * 512 + 255) / 256, 256, 0, stream>>>(gat_w1, w1frag, 96, 64, 0);
  k_prep_wfrag<<<(2 * 2 * 512 + 255) / 256, 256, 0, stream>>>(gat_w2, w2frag, 64, 32, 0);

  // ---- conv branch: im2col GEMM (M=24576, K=21504, N=64) + bias/leaky/BN ----
  k_gemm_bf16<<<N_ / 16, 128, 0, stream>>>(t5pad, DT5_, L_, 20 * DT5_, convfrag, KCONV_,
                                           nullptr, 0, z, zbf, 32, conv_b,
                                           bn_g, bn_b, bn_m, bn_v, 0);

  // ---- biLSTM over local windows -> hcat bf16 (N, 352) ----
  k_lstm<<<(2 * N_) / 256, 256, 0, stream>>>(feature, ln_g, ln_b,
                                             wih_f, whh_f, bih_f, bhh_f,
                                             wih_b, whh_b, bih_b, bhh_b, hcat);

  // ---- dense GEMM (M=24576, K=352, N=32) + bias/relu -> z[:,0:32] ----
  k_gemm_bf16<<<N_ / 16, 64, 0, stream>>>(hcat, 352, 1 << 30, 0, densefrag, 352,
                                          nullptr, 0, z, zbf, 0, dense_b,
                                          nullptr, nullptr, nullptr, nullptr, 1);

  // ---- GAT layer 1 ----
  k_gemm_bf16<<<N_ / 16, 128, 0, stream>>>(zbf, DZ_, 1 << 30, 0, w1frag, 96,
                                           z1, 64, nullptr, nullptr, 0, nullptr,
                                           nullptr, nullptr, nullptr, nullptr, 2);
  k_edge_score<<<E_ / 256, 256, 0, stream>>>(z1, 64, src, dst, edgef, gat_we1, gat_a1,
                                             score1, segmax1);
  k_edge_norm<<<E_ / 256, 256, 0, stream>>>(score1, dst, segmax1, ex1, segsum1);
  k_edge_aggr<<<E_ / 256, 256, 0, stream>>>(z1, 64, src, dst, edgef, gat_we1,
                                            ex1, segsum1, h1, nullptr);
  k_elu_bf<<<(N_ * 64) / 256, 256, 0, stream>>>(h1, h1bf, (long)N_ * 64);

  // ---- GAT layer 2 ----
  k_gemm_bf16<<<N_ / 16, 64, 0, stream>>>(h1bf, 64, 1 << 30, 0, w2frag, 64,
                                          z2, 32, nullptr, nullptr, 0, nullptr,
                                          nullptr, nullptr, nullptr, nullptr, 2);
  k_edge_score<<<E_ / 256, 256, 0, stream>>>(z2, 32, src, dst, edgef, gat_we2, gat_a2,
                                             score2, segmax2);
  k_edge_norm<<<E_ / 256, 256, 0, stream>>>(score2, dst, segmax2, ex2, segsum2);
  k_edge_aggr<<<E_ / 256, 256, 0, stream>>>(z2, 32, src, dst, edgef, gat_we2,
                                            ex2, segsum2, h2, alpha2);

  // ---- head: sigmoid(concat(h2, z) @ out_w^T + b) ----
  k_out_final<<<N_ / 256, 256, 0, stream>>>(h2, z, out_w, out_b, out);

  (void)in_sizes; (void)n_in; (void)out_size; (void)ws_size;
}
